// MultiRNN_47158740910641
// MI455X (gfx1250) — compile-verified, matrix operand order checked by reading, not run on hardware
//
#include <hip/hip_runtime.h>
#include <hip/hip_bf16.h>
#include <math.h>

// ---------------- problem constants ----------------
#define NS    131072     // samples per event
#define NF    128        // frames
#define WIN   1024       // window
#define CPD   16         // control plane dim
#define NCP   512
#define NV    8
#define NVERB 16
#define HID   128
#define SPARS 128        // top-k
#define NTOT  128        // B*E
#define NFFT  262144     // 2*NS, power of two (2^18)
#define LOG2NS 17        // NS = 2^17
#define LOG2NFFT 18
#define CHUNK 16         // events per FFT chunk (== NVERB, shared batch kernels)

typedef __attribute__((ext_vector_type(16))) _Float16 v16h;
typedef __attribute__((ext_vector_type(8)))  _Float16 v8h;
typedef __attribute__((ext_vector_type(8)))  float    v8f;
typedef __attribute__((ext_vector_type(2)))  float    v2f;
typedef __attribute__((ext_vector_type(4)))  unsigned int u32x4;
typedef __attribute__((ext_vector_type(8)))  int      i32x8;
typedef __attribute__((ext_vector_type(4)))  int      i32x4;

// ---------------------------------------------------------------------------
// K1: per-event argmax routing + softmax(mix) + |amp|
// ---------------------------------------------------------------------------
__device__ __forceinline__ int dargmax(const float* p, int n) {
    int bi = 0; float bv = p[0];
    for (int i = 1; i < n; ++i) { float v = p[i]; if (v > bv) { bv = v; bi = i; } }
    return bi;
}

__global__ void select_kernel(const float* voice, const float* cpc, const float* amps,
                              const float* room, const float* rmix, const float* times,
                              int* selV, int* selCP, int* selR, int* selT,
                              float* mix0, float* mix1, float* amp) {
    int n = blockIdx.x * blockDim.x + threadIdx.x;
    if (n >= NTOT) return;
    selV[n]  = dargmax(voice + (size_t)n * NV,   NV);
    selCP[n] = dargmax(cpc   + (size_t)n * NCP,  NCP);
    selR[n]  = dargmax(room  + (size_t)n * NVERB, NVERB);
    selT[n]  = dargmax(times + (size_t)n * NF,   NF);
    float a = rmix[2 * n], b = rmix[2 * n + 1];
    float m = fmaxf(a, b);
    float ea = __expf(a - m), eb = __expf(b - m), inv = 1.f / (ea + eb);
    mix0[n] = ea * inv;   // wet coefficient
    mix1[n] = eb * inv;   // dry coefficient
    amp[n]  = fabsf(amps[n]);
}

// ---------------------------------------------------------------------------
// K2a: w_out f32 -> f16
// ---------------------------------------------------------------------------
__global__ void cvt_wout_kernel(const float* w, _Float16* o, int n) {
    int i = blockIdx.x * blockDim.x + threadIdx.x;
    if (i < n) o[i] = (_Float16)w[i];
}

// ---------------------------------------------------------------------------
// K2b: per-voice Whh -> transposed f16 (whhT[v][k*HID+h] = w_hh[v][h][k]),
//      so the RNN kernel's LDS image is a flat contiguous 32KB block (TDM-able)
//      and the per-step reads are LDS-bank-conflict free.
// ---------------------------------------------------------------------------
__global__ void cvt_whh_kernel(const float* w_hh, _Float16* whhT) {
    int i = blockIdx.x * blockDim.x + threadIdx.x;      // < NV*HID*HID
    int v = i >> 14, rem = i & (HID * HID - 1);
    int h = rem >> 7, k = rem & (HID - 1);
    whhT[(size_t)v * HID * HID + k * HID + h] = (_Float16)w_hh[i];
}

// ---------------------------------------------------------------------------
// K3: control plane row gather + top-128 sparsify (bisection threshold) + relu
//     src row layout [CPD][NF] -> ctrl layout [NF][CPD]
// ---------------------------------------------------------------------------
__global__ void __launch_bounds__(256) ctrl_kernel(const float* cp_table, const int* selCP,
                                                   float* ctrl) {
    __shared__ float vals[NF * CPD];   // 2048
    __shared__ float red[256];
    __shared__ int   ired[256];
    int e = blockIdx.x, tid = threadIdx.x;
    const float* row = cp_table + (size_t)selCP[e] * (NF * CPD);
    for (int i = tid; i < NF * CPD; i += 256) vals[i] = row[i];
    __syncthreads();

    // min / max reductions
    float mx = -1e30f, mn = 1e30f;
    for (int i = tid; i < NF * CPD; i += 256) { float x = vals[i]; mx = fmaxf(mx, x); mn = fminf(mn, x); }
    red[tid] = mx; __syncthreads();
    for (int s = 128; s > 0; s >>= 1) { if (tid < s) red[tid] = fmaxf(red[tid], red[tid + s]); __syncthreads(); }
    mx = red[0]; __syncthreads();
    red[tid] = mn; __syncthreads();
    for (int s = 128; s > 0; s >>= 1) { if (tid < s) red[tid] = fminf(red[tid], red[tid + s]); __syncthreads(); }
    mn = red[0]; __syncthreads();

    // bisect threshold t: count(v > t) converges to SPARS
    float lo = mn - 1e-6f, hi = mx;
    for (int it = 0; it < 32; ++it) {
        float mid = 0.5f * (lo + hi);
        int c = 0;
        for (int i = tid; i < NF * CPD; i += 256) c += (vals[i] > mid) ? 1 : 0;
        ired[tid] = c; __syncthreads();
        for (int s = 128; s > 0; s >>= 1) { if (tid < s) ired[tid] += ired[tid + s]; __syncthreads(); }
        int cnt = ired[0];
        __syncthreads();
        if (cnt > SPARS) lo = mid; else hi = mid;
    }
    float thr = hi;
    for (int i = tid; i < NF * CPD; i += 256) {
        int c = i >> 7;        // i = c*NF + f
        int f = i & (NF - 1);
        float x = vals[i];
        ctrl[(size_t)e * (NF * CPD) + f * CPD + c] = (x > thr) ? fmaxf(x, 0.f) : 0.f;
    }
}

// ---------------------------------------------------------------------------
// K4: per-voice M = w_ih(128x1024) @ w_in(1024x16)  via v_wmma_f32_16x16x4_f32
//     one wave per (voice, m-tile); K-loop of 256 WMMAs
// ---------------------------------------------------------------------------
__global__ void __launch_bounds__(32) gemm_M_kernel(const float* w_ih, const float* w_in,
                                                    float* Mmat) {
    int v = blockIdx.y, mt = blockIdx.x;
    int lane = threadIdx.x;
    int hf = lane >> 4, r = lane & 15;
    const float* Ap = w_ih + (size_t)v * HID * WIN + (size_t)(mt * 16 + r) * WIN;
    const float* Bp = w_in + (size_t)v * WIN * CPD;
    v8f acc = {0.f, 0.f, 0.f, 0.f, 0.f, 0.f, 0.f, 0.f};
    for (int k0 = 0; k0 < WIN; k0 += 4) {
        v2f a = *(const v2f*)(Ap + k0 + 2 * hf);            // A[m][k], A[m][k+1]
        v2f b;                                              // B[k][n] = w_in[k][n]
        b.x = Bp[(size_t)(k0 + 2 * hf) * CPD + r];
        b.y = Bp[(size_t)(k0 + 2 * hf + 1) * CPD + r];
        acc = __builtin_amdgcn_wmma_f32_16x16x4_f32(false, a, false, b, (short)0, acc, false, false);
    }
#pragma unroll
    for (int rr = 0; rr < 8; ++rr) {
        int m = mt * 16 + rr + 8 * hf;
        Mmat[(size_t)v * HID * CPD + (size_t)m * CPD + r] = acc[rr];
    }
}

// ---------------------------------------------------------------------------
// K5: per-event A = ctrl(128x16) @ M^T(16x128)  via v_wmma_f32_16x16x4_f32
//     block = event (8 waves = 8 n-tiles, loop over 8 m-tiles)
// ---------------------------------------------------------------------------
__global__ void __launch_bounds__(256) gemm_A_kernel(const float* ctrl, const float* Mmat,
                                                     const int* selV, float* Amat) {
    int e = blockIdx.x;
    int wave = threadIdx.x >> 5, lane = threadIdx.x & 31;
    int hf = lane >> 4, r = lane & 15;
    int nt = wave;
    const float* C  = ctrl + (size_t)e * NF * CPD;
    const float* Mv = Mmat + (size_t)selV[e] * HID * CPD;
    for (int mt = 0; mt < 8; ++mt) {
        v8f acc = {0.f, 0.f, 0.f, 0.f, 0.f, 0.f, 0.f, 0.f};
#pragma unroll
        for (int k0 = 0; k0 < CPD; k0 += 4) {
            v2f a = *(const v2f*)(C  + (size_t)(mt * 16 + r) * CPD + k0 + 2 * hf);
            v2f b = *(const v2f*)(Mv + (size_t)(nt * 16 + r) * CPD + k0 + 2 * hf); // B[k][n]=M[n][k]
            acc = __builtin_amdgcn_wmma_f32_16x16x4_f32(false, a, false, b, (short)0, acc, false, false);
        }
#pragma unroll
        for (int rr = 0; rr < 8; ++rr) {
            int m = mt * 16 + rr + 8 * hf;
            Amat[(size_t)e * NF * HID + (size_t)m * HID + nt * 16 + r] = acc[rr];
        }
    }
}

// ---------------------------------------------------------------------------
// K6: sequential recurrence  h_t = tanh(A_t + Whh h_{t-1});  writes hs as f16.
//     Whh^T (pre-transposed f16, 32KB) pulled into LDS by one Tensor-Data-Mover
//     DMA issued by wave 0, synchronized via s_wait_tensorcnt + barrier.
// ---------------------------------------------------------------------------
__global__ void __launch_bounds__(128) rnn_kernel(const _Float16* whhT_g, const float* Amat,
                                                  const int* selV, _Float16* hs) {
    __shared__ _Float16 whhT[HID * HID];  // 32 KB, flat image of whhT_g[v]
    __shared__ float hbuf[2][HID];
    int e = blockIdx.x, tid = threadIdx.x;
    const _Float16* W = whhT_g + (size_t)selV[e] * HID * HID;
    hbuf[0][tid] = 0.f;

#if defined(__has_builtin)
#if __has_builtin(__builtin_amdgcn_tensor_load_to_lds) && __has_builtin(__builtin_amdgcn_s_wait_tensorcnt)
#define USED_TDM 1
    if (tid < 32) {   // wave 0 issues a single TDM DMA: 32KB global -> LDS
        unsigned lds_off = (unsigned)(size_t)(void*)&whhT[0];     // shared aperture: low 32b = LDS offset
        unsigned long long ga = (unsigned long long)(size_t)W;    // 57-bit global address
        // ---- D# group 0 (128b): count=1 | lds_addr | global_addr | type=2 ----
        u32x4 g0;
        g0[0] = 1u;                                               // count=1, user mode
        g0[1] = lds_off;                                          // lds_addr (bytes)
        g0[2] = (unsigned)ga;                                     // global_addr[31:0]
        g0[3] = (unsigned)((ga >> 32) & 0x01FFFFFFu) | (2u << 30);// global_addr[56:32] | type=2
        // ---- D# group 1 (256b): 1-row 2D tile, data_size=2B, 16384 elements ----
        i32x8 g1;
        g1[0] = 0x10000;                       // workgroup_mask=0, data_size=1 (2 bytes)
        g1[1] = (int)((HID * HID) << 16);      // tensor_dim0[15:0] in [31:16]
        g1[2] = (int)(1 << 16);                // tensor_dim0[31:16]=0 | tensor_dim1[15:0]=1
        g1[3] = (int)((HID * HID) << 16);      // tensor_dim1[31:16]=0 | tile_dim0=16384
        g1[4] = 1;                             // tile_dim1=1, tile_dim2=0
        g1[5] = HID * HID;                     // tensor_dim0_stride[31:0]
        g1[6] = 0;                             // stride hi | tensor_dim1_stride lo
        g1[7] = 0;
        i32x4 gz4 = {0, 0, 0, 0};              // groups 2/3 unused (<=2D tensor)
        i32x8 gz8 = {0, 0, 0, 0, 0, 0, 0, 0};  // trailing descriptor group (unused)
        __builtin_amdgcn_tensor_load_to_lds(g0, g1, gz4, gz4, gz8, 0);
        __builtin_amdgcn_s_wait_tensorcnt((short)0);              // per-wave TENSORcnt
    }
#endif
#endif
#ifndef USED_TDM
    for (int i = tid; i < HID * HID; i += 128) whhT[i] = W[i];    // fallback staging
#endif
    __syncthreads();   // cross-wave handshake: LDS image complete

    const float* Ae = Amat + (size_t)e * NF * HID;
    _Float16* He = hs + (size_t)e * NF * HID;
    int cur = 0;
    for (int t = 0; t < NF; ++t) {
        float acc = Ae[(size_t)t * HID + tid];
        if (t + 1 < NF) __builtin_prefetch(&Ae[(size_t)(t + 1) * HID + tid], 0, 0);
        const float* h = hbuf[cur];
#pragma unroll 8
        for (int k = 0; k < HID; ++k) acc += (float)whhT[k * HID + tid] * h[k];
        float hn = tanhf(acc);
        hbuf[cur ^ 1][tid] = hn;
        He[(size_t)t * HID + tid] = (_Float16)hn;
        cur ^= 1;
        __syncthreads();
    }
}

// ---------------------------------------------------------------------------
// K7: sig = sin(hs(128x128) @ w_out^T(128x1024))  via v_wmma_f32_16x16x32_f16
//     grid (8 n-groups, 8 m-tiles, events); wave = one 16x16 tile, K=128 in 4 steps
// ---------------------------------------------------------------------------
__global__ void __launch_bounds__(256) gemm_sig_kernel(const _Float16* hs, const _Float16* woutH,
                                                       const int* selV, float* sig) {
    int e = blockIdx.z, mt = blockIdx.y;
    int wave = threadIdx.x >> 5, lane = threadIdx.x & 31;
    int nt = blockIdx.x * 8 + wave;
    int hf = lane >> 4, r = lane & 15;
    const _Float16* Ap = hs    + (size_t)e * NF * HID        + (size_t)(mt * 16 + r) * HID;
    const _Float16* Bp = woutH + (size_t)selV[e] * WIN * HID + (size_t)(nt * 16 + r) * HID;
    v8f acc = {0.f, 0.f, 0.f, 0.f, 0.f, 0.f, 0.f, 0.f};
#pragma unroll
    for (int k0 = 0; k0 < HID; k0 += 32) {
        int base = k0 + 8 * hf;                    // ISA 16-bit A/B fragment K packing
        v8h al = *(const v8h*)(Ap + base);
        v8h ah = *(const v8h*)(Ap + base + 16);
        v8h bl = *(const v8h*)(Bp + base);
        v8h bh = *(const v8h*)(Bp + base + 16);
        v16h a, b;
#pragma unroll
        for (int i = 0; i < 8; ++i) { a[i] = al[i]; a[i + 8] = ah[i]; b[i] = bl[i]; b[i + 8] = bh[i]; }
        acc = __builtin_amdgcn_wmma_f32_16x16x32_f16(false, a, false, b, (short)0, acc, false, false);
    }
    float* o = sig + (size_t)e * NS;
#pragma unroll
    for (int rr = 0; rr < 8; ++rr) {
        int m = mt * 16 + rr + 8 * hf;
        o[(size_t)m * WIN + nt * 16 + r] = __sinf(acc[rr]);
    }
}

// ---------------------------------------------------------------------------
// FFT convolution machinery: in-place DIF forward (natural -> bit-reversed),
// bit-reversed pointwise multiply, in-place DIT inverse (bit-reversed -> natural).
// Batch of 16 transforms per launch (== NVERB == CHUNK).
// ---------------------------------------------------------------------------
__global__ void verb_pad_kernel(const float* verbs, float2* VF) {
    int gid = blockIdx.x * blockDim.x + threadIdx.x;    // < NVERB*NFFT
    int ir = gid >> LOG2NFFT, i = gid & (NFFT - 1);
    float v = (i < NS) ? verbs[(size_t)ir * NS + i] : 0.f;
    VF[(size_t)ir * NFFT + i] = make_float2(v, 0.f);
}

__global__ void pad_sig_kernel(const float* sig, float2* X, int e0) {
    int gid = blockIdx.x * blockDim.x + threadIdx.x;    // < CHUNK*NFFT
    int ce = gid >> LOG2NFFT, i = gid & (NFFT - 1);
    float v = (i < NS) ? sig[(size_t)(e0 + ce) * NS + i] : 0.f;
    X[(size_t)ce * NFFT + i] = make_float2(v, 0.f);
}

__global__ void fft_dif_pass(float2* d, int len) {
    int gid = blockIdx.x * blockDim.x + threadIdx.x;    // < 16 * NFFT/2
    int ev = gid >> (LOG2NFFT - 1);
    int b  = gid & ((NFFT >> 1) - 1);
    int hl = len >> 1;
    int blk = b / hl, k = b - blk * hl;
    size_t j = (size_t)ev * NFFT + (size_t)blk * len + k;
    float2 u = d[j], v = d[j + hl];
    float ang = -6.283185307179586f * (float)k / (float)len;
    float sn, cs; __sincosf(ang, &sn, &cs);
    float tx = u.x - v.x, ty = u.y - v.y;
    d[j]      = make_float2(u.x + v.x, u.y + v.y);
    d[j + hl] = make_float2(tx * cs - ty * sn, tx * sn + ty * cs);
}

__global__ void fft_dit_pass(float2* d, int len) {
    int gid = blockIdx.x * blockDim.x + threadIdx.x;    // < 16 * NFFT/2
    int ev = gid >> (LOG2NFFT - 1);
    int b  = gid & ((NFFT >> 1) - 1);
    int hl = len >> 1;
    int blk = b / hl, k = b - blk * hl;
    size_t j = (size_t)ev * NFFT + (size_t)blk * len + k;
    float2 u = d[j], v = d[j + hl];
    float ang = 6.283185307179586f * (float)k / (float)len;
    float sn, cs; __sincosf(ang, &sn, &cs);
    float tx = v.x * cs - v.y * sn, ty = v.x * sn + v.y * cs;
    d[j]      = make_float2(u.x + tx, u.y + ty);
    d[j + hl] = make_float2(u.x - tx, u.y - ty);
}

__global__ void specmul_kernel(float2* X, const float2* VF, const int* selR, int e0) {
    int gid = blockIdx.x * blockDim.x + threadIdx.x;    // < CHUNK*NFFT
    int ce = gid >> LOG2NFFT, i = gid & (NFFT - 1);
    int r = selR[e0 + ce];
    float2 a = X[(size_t)ce * NFFT + i];
    float2 b = VF[(size_t)r * NFFT + i];
    X[(size_t)ce * NFFT + i] = make_float2(a.x * b.x - a.y * b.y, a.x * b.y + a.y * b.x);
}

// ---------------------------------------------------------------------------
// K-final: out[s] = (mix0*wet[s-d] + mix1*sig[s-d]) * |amp|,  d = argmax(times)*WIN
// ---------------------------------------------------------------------------
__global__ void finalize_kernel(float* out, const float2* X, const float* sig,
                                const int* selT, const float* mix0, const float* mix1,
                                const float* amp, int e0) {
    int gid = blockIdx.x * blockDim.x + threadIdx.x;    // < CHUNK*NS
    int ce = gid >> LOG2NS, s = gid & (NS - 1);
    int e = e0 + ce;
    int d = selT[e] * WIN;
    float v = 0.f;
    if (s >= d) {
        int sd = s - d;
        float wet = X[(size_t)ce * NFFT + sd].x * (1.f / (float)NFFT);
        float dry = sig[(size_t)e * NS + sd];
        v = (mix0[e] * wet + mix1[e] * dry) * amp[e];
    }
    out[(size_t)e * NS + s] = v;
}

// ---------------------------------------------------------------------------
// host launcher
// ---------------------------------------------------------------------------
extern "C" void kernel_launch(void* const* d_in, const int* in_sizes, int n_in,
                              void* d_out, int out_size, void* d_ws, size_t ws_size,
                              hipStream_t stream) {
    (void)in_sizes; (void)n_in; (void)out_size; (void)ws_size;
    const float* voice    = (const float*)d_in[0];
    const float* cpc      = (const float*)d_in[1];
    const float* amps     = (const float*)d_in[2];
    const float* room     = (const float*)d_in[3];
    const float* rmix     = (const float*)d_in[4];
    const float* times    = (const float*)d_in[5];
    const float* cp_table = (const float*)d_in[6];
    const float* verbs    = (const float*)d_in[7];
    const float* w_in     = (const float*)d_in[8];
    const float* w_ih     = (const float*)d_in[9];
    const float* w_hh     = (const float*)d_in[10];
    const float* w_out    = (const float*)d_in[11];
    float* out = (float*)d_out;

    // workspace bump allocator (~151 MB total)
    char* ws = (char*)d_ws;
    size_t off = 0;
    auto alloc = [&](size_t bytes) -> void* {
        void* p = ws + off;
        off = (off + bytes + 255) & ~(size_t)255;
        return p;
    };
    int*   selV  = (int*)alloc(NTOT * sizeof(int));
    int*   selCP = (int*)alloc(NTOT * sizeof(int));
    int*   selR  = (int*)alloc(NTOT * sizeof(int));
    int*   selT  = (int*)alloc(NTOT * sizeof(int));
    float* mix0  = (float*)alloc(NTOT * sizeof(float));
    float* mix1  = (float*)alloc(NTOT * sizeof(float));
    float* amp   = (float*)alloc(NTOT * sizeof(float));
    float* Mmat  = (float*)alloc((size_t)NV * HID * CPD * sizeof(float));
    float* ctrl  = (float*)alloc((size_t)NTOT * NF * CPD * sizeof(float));
    float* Amat  = (float*)alloc((size_t)NTOT * NF * HID * sizeof(float));
    _Float16* hsH    = (_Float16*)alloc((size_t)NTOT * NF * HID * sizeof(_Float16));
    _Float16* woutH  = (_Float16*)alloc((size_t)NV * WIN * HID * sizeof(_Float16));
    _Float16* whhTH  = (_Float16*)alloc((size_t)NV * HID * HID * sizeof(_Float16));
    float*  sig = (float*)alloc((size_t)NTOT * NS * sizeof(float));
    float2* VF  = (float2*)alloc((size_t)NVERB * NFFT * sizeof(float2));
    float2* X   = (float2*)alloc((size_t)CHUNK * NFFT * sizeof(float2));

    // routing / scalars
    select_kernel<<<1, 128, 0, stream>>>(voice, cpc, amps, room, rmix, times,
                                         selV, selCP, selR, selT, mix0, mix1, amp);
    // f16 weight conversions (w_out flat; w_hh transposed per voice for LDS image)
    cvt_wout_kernel<<<(NV * WIN * HID) / 256, 256, 0, stream>>>(w_out, woutH, NV * WIN * HID);
    cvt_whh_kernel<<<(NV * HID * HID) / 256, 256, 0, stream>>>(w_hh, whhTH);
    // control plane gather + top-k + relu
    ctrl_kernel<<<NTOT, 256, 0, stream>>>(cp_table, selCP, ctrl);
    // per-voice fused input matrix  M = w_ih @ w_in   (f32 WMMA)
    gemm_M_kernel<<<dim3(HID / 16, NV), 32, 0, stream>>>(w_ih, w_in, Mmat);
    // per-event RNN input  A = ctrl @ M^T   (f32 WMMA)
    gemm_A_kernel<<<NTOT, 256, 0, stream>>>(ctrl, Mmat, selV, Amat);
    // sequential recurrence -> hs (f16); Whh^T staged via Tensor Data Mover
    rnn_kernel<<<NTOT, 128, 0, stream>>>(whhTH, Amat, selV, hsH);
    // sig = sin(hs @ w_out^T)   (f16 WMMA)
    gemm_sig_kernel<<<dim3(8, 8, NTOT), 256, 0, stream>>>(hsH, woutH, selV, sig);

    // verb spectra (bit-reversed order), recomputed each call for determinism
    verb_pad_kernel<<<(NVERB * NFFT) / 256, 256, 0, stream>>>(verbs, VF);
    for (int len = NFFT; len >= 2; len >>= 1)
        fft_dif_pass<<<(NVERB * (NFFT / 2)) / 256, 256, 0, stream>>>(VF, len);

    // reverb + mix + amplitude + dirac shift, 8 chunks of 16 events
    for (int c = 0; c < NTOT / CHUNK; ++c) {
        int e0 = c * CHUNK;
        pad_sig_kernel<<<(CHUNK * NFFT) / 256, 256, 0, stream>>>(sig, X, e0);
        for (int len = NFFT; len >= 2; len >>= 1)
            fft_dif_pass<<<(CHUNK * (NFFT / 2)) / 256, 256, 0, stream>>>(X, len);
        specmul_kernel<<<(CHUNK * NFFT) / 256, 256, 0, stream>>>(X, VF, selR, e0);
        for (int len = 2; len <= NFFT; len <<= 1)
            fft_dit_pass<<<(CHUNK * (NFFT / 2)) / 256, 256, 0, stream>>>(X, len);
        finalize_kernel<<<(CHUNK * NS) / 256, 256, 0, stream>>>(out, X, sig, selT,
                                                                mix0, mix1, amp, e0);
    }
}
